// CausalSelfAttention_8993661517830
// MI455X (gfx1250) — compile-verified, matrix operand order checked by reading, not run on hardware
//
#include <hip/hip_runtime.h>

// ---------------------------------------------------------------------------
// CDNA5 (gfx1250) causal self-attention, bf16 WMMA + Tensor Data Mover.
//   B=2, T=2048, C=2048, H=16, D=128.
//   cvt f32->bf16 (x, W*) -> q/k/v = x@W.T (TDM-staged bf16 WMMA, bf16 out)
//   -> qk-norm+RoPE (bf16 in/out, f32 math) -> flash attention (TDM K-tile,
//   bf16 WMMA, online softmax via ds_swizzle butterflies) -> y@Wo.T (f32).
// ---------------------------------------------------------------------------

typedef __attribute__((ext_vector_type(16))) __bf16    v16bf;
typedef __attribute__((ext_vector_type(8)))  float     v8f;
typedef __attribute__((ext_vector_type(4)))  unsigned  v4u;
typedef __attribute__((ext_vector_type(8)))  int       v8i;
typedef __attribute__((ext_vector_type(4)))  int       v4i;

union U16x { v16bf v; __bf16 u[16]; uint4 q[2]; };

__device__ __forceinline__ v8f wmma_bf16(v16bf a, v16bf b, v8f c) {
  return __builtin_amdgcn_wmma_f32_16x16x32_bf16(
      false, a, false, b, (short)0, c, false, false);
}

// xor-butterfly via ds_swizzle (group-of-32: imm = xor<<10 | or<<5 | and)
template <int IMM>
__device__ __forceinline__ float swz(float x) {
  union { float f; int i; } u; u.f = x;
  u.i = __builtin_amdgcn_ds_swizzle(u.i, IMM);
  return u.f;
}
__device__ __forceinline__ float redmax16(float x) {
  x = fmaxf(x, swz<0x041F>(x));   // xor 1
  x = fmaxf(x, swz<0x081F>(x));   // xor 2
  x = fmaxf(x, swz<0x101F>(x));   // xor 4
  x = fmaxf(x, swz<0x201F>(x));   // xor 8  (stays within each 16-lane half)
  return x;
}
__device__ __forceinline__ float redsum16(float x) {
  x += swz<0x041F>(x);
  x += swz<0x081F>(x);
  x += swz<0x101F>(x);
  x += swz<0x201F>(x);
  return x;
}

#if __has_builtin(__builtin_amdgcn_tensor_load_to_lds) && \
    __has_builtin(__builtin_amdgcn_s_wait_tensorcnt)
#define USE_TDM 1
#else
#define USE_TDM 0
#endif

#if USE_TDM
// 2D bf16 tile load, global -> LDS, with per-row LDS padding.
__device__ __forceinline__ void tdm_load_2d(
    unsigned lds_off, const void* gaddr,
    unsigned tile_d0, unsigned tile_d1, unsigned row_stride,
    unsigned pad_i, unsigned pad_a) {
  unsigned long long ga = (unsigned long long)(size_t)gaddr;
  v4u g0;
  g0.x = 1u;                                   // count=1 (valid), user mode
  g0.y = lds_off;                              // LDS byte address
  g0.z = (unsigned)ga;                         // global addr [31:0]
  g0.w = (unsigned)(ga >> 32) | (2u << 30);    // global addr [56:32] | type=2
  v8i g1;
  g1[0] = (int)((1u << 16) | (1u << 20) | (pad_i << 22) | (pad_a << 25));
  //            data_size=2B  pad_enable   pad_interval    pad_amount
  g1[1] = (int)((tile_d0 & 0xFFFFu) << 16);            // tensor_dim0 lo16
  g1[2] = (int)((tile_d0 >> 16) | ((tile_d1 & 0xFFFFu) << 16)); // d0 hi | d1 lo
  g1[3] = (int)((tile_d1 >> 16) | (tile_d0 << 16));    // d1 hi | tile_dim0
  g1[4] = (int)(tile_d1 & 0xFFFFu);                    // tile_dim1 (dim2 = 0)
  g1[5] = (int)row_stride;                             // tensor_dim0_stride lo32
  g1[6] = 0;
  g1[7] = 0;
  v4i z4 = {0, 0, 0, 0};
  v8i z8 = {0, 0, 0, 0, 0, 0, 0, 0};
  __builtin_amdgcn_tensor_load_to_lds(g0, g1, z4, z4, z8, 0);
}
#endif

// ---------------------------------------------------------------------------
// f32 -> bf16 bulk convert (n multiple of 4)
// ---------------------------------------------------------------------------
__global__ __launch_bounds__(256) void cvt_f32_bf16(
    const float* __restrict__ s, __bf16* __restrict__ d, int n) {
  int i = (blockIdx.x * 256 + threadIdx.x) * 4;
  if (i + 3 < n) {
    float4 f = *(const float4*)(s + i);
    d[i + 0] = (__bf16)f.x; d[i + 1] = (__bf16)f.y;
    d[i + 2] = (__bf16)f.z; d[i + 3] = (__bf16)f.w;
  }
}

// ---------------------------------------------------------------------------
// NT GEMM: C[M,2048] = A[M,2048] * B[2048,2048]^T, A/B bf16 in memory.
// ---------------------------------------------------------------------------
#define GS 40   // LDS row stride in bf16 (32 + 8 pad) -> conflict-free

template <bool OUT_F32>
__global__ __launch_bounds__(256) void gemm_nt_bf16(
    const __bf16* __restrict__ A, const __bf16* __restrict__ B,
    void* __restrict__ Cout) {
  constexpr int NK = 2048;
  __shared__ __bf16 sa[128 * GS];
  __shared__ __bf16 sb[128 * GS];

  const int tid  = threadIdx.x;
  const int lane = tid & 31, wave = tid >> 5;
  const int wm = wave & 1, wn = wave >> 1;
  const int lm = lane & 15, hi = lane >> 4;
  const int kb = hi ? 8 : 0;
  const int bm = blockIdx.y * 128, bn = blockIdx.x * 128;

  v8f acc[4][2] = {};

#if USE_TDM
  const unsigned sa_off = (unsigned)(size_t)(void*)sa;
  const unsigned sb_off = (unsigned)(size_t)(void*)sb;
#endif

  for (int k0 = 0; k0 < NK; k0 += 32) {
    __syncthreads();
#if USE_TDM
    if (wave == 0) {
      tdm_load_2d(sa_off, A + (size_t)bm * NK + k0, 32u, 128u, NK, 3u, 3u);
      tdm_load_2d(sb_off, B + (size_t)bn * NK + k0, 32u, 128u, NK, 3u, 3u);
      __builtin_amdgcn_s_wait_tensorcnt((short)0);
    }
#else
#pragma unroll
    for (int i = 0; i < 2; ++i) {
      int row = (tid >> 2) + i * 64;
      int cs  = (tid & 3) * 8;
      *(uint4*)&sa[row * GS + cs] =
          *(const uint4*)(A + (size_t)(bm + row) * NK + k0 + cs);
      *(uint4*)&sb[row * GS + cs] =
          *(const uint4*)(B + (size_t)(bn + row) * NK + k0 + cs);
    }
#endif
    __syncthreads();

    v16bf af[4], bf[2];
#pragma unroll
    for (int i = 0; i < 4; ++i) {
      int row = wm * 64 + i * 16 + lm;
      U16x t;
      t.q[0] = *(const uint4*)&sa[row * GS + kb];
      t.q[1] = *(const uint4*)&sa[row * GS + kb + 16];
      af[i] = t.v;
    }
#pragma unroll
    for (int j = 0; j < 2; ++j) {
      int row = wn * 32 + j * 16 + lm;
      U16x t;
      t.q[0] = *(const uint4*)&sb[row * GS + kb];
      t.q[1] = *(const uint4*)&sb[row * GS + kb + 16];
      bf[j] = t.v;
    }
#pragma unroll
    for (int i = 0; i < 4; ++i)
#pragma unroll
      for (int j = 0; j < 2; ++j)
        acc[i][j] = wmma_bf16(af[i], bf[j], acc[i][j]);
  }

  const int r0 = bm + wm * 64 + 8 * hi;
  const int c0 = bn + wn * 32 + lm;
#pragma unroll
  for (int i = 0; i < 4; ++i)
#pragma unroll
    for (int j = 0; j < 2; ++j)
#pragma unroll
      for (int v = 0; v < 8; ++v) {
        size_t idx = (size_t)(r0 + i * 16 + v) * NK + c0 + j * 16;
        if (OUT_F32) ((float*)Cout)[idx] = acc[i][j][v];
        else         ((__bf16*)Cout)[idx] = (__bf16)acc[i][j][v];
      }
}

// ---------------------------------------------------------------------------
// Per-(b,t,h) L2 norm + qk_scale*sqrt(C) + RoPE, bf16 in/out, f32 math.
// ---------------------------------------------------------------------------
__global__ __launch_bounds__(128) void normrope(
    __bf16* __restrict__ q, __bf16* __restrict__ k,
    const float* __restrict__ qk_scale, int T) {
  const int h = blockIdx.x, t = blockIdx.y, b = blockIdx.z;
  const int d = threadIdx.x;
  __shared__ float sv[128];
  __shared__ float red[4];

  const size_t base = ((size_t)(b * T + t)) * 2048 + h * 128;
  const float scale = qk_scale[d] * 45.254833995939045f;  // sqrt(2048)

  for (int which = 0; which < 2; ++which) {
    __bf16* p = which ? k : q;
    float val = (float)p[base + d];
    float ss = val * val;
    ss = redsum16(ss);
    ss += swz<0x401F>(ss);   // xor 16: full 32-lane sum
    if ((d & 31) == 0) red[d >> 5] = ss;
    __syncthreads();
    float nrm = sqrtf(red[0] + red[1] + red[2] + red[3]);
    sv[d] = val * scale / fmaxf(nrm, 1e-12f);
    __syncthreads();
    if (d < 64) {
      float inv = __powf(10000.0f, -(float)(2 * d) / 128.0f);
      float ang = (float)t * inv;
      float c = __cosf(ang), s = __sinf(ang);
      float x1 = sv[d], x2 = sv[d + 64];
      p[base + d]      = (__bf16)( x1 * c + x2 * s);
      p[base + d + 64] = (__bf16)(-x1 * s + x2 * c);
    }
    __syncthreads();
  }
}

// ---------------------------------------------------------------------------
// Flash attention: grid = (T/128, H, B), 256 threads (8 waves x 16 q-rows).
// ---------------------------------------------------------------------------
#define KTS 136   // kt row stride (128 + 8)
#define VTS 40    // vt / pb row stride (32 + 8)

__global__ __launch_bounds__(256) void flash_attn(
    const __bf16* __restrict__ Q, const __bf16* __restrict__ K,
    const __bf16* __restrict__ V, __bf16* __restrict__ Y, int T) {
  __shared__ __bf16 kt[32 * KTS];     // [token][d]
  __shared__ __bf16 vt[128 * VTS];    // [d][token]  (transposed V)
  __shared__ __bf16 pb[8 * 16 * VTS]; // per-wave P bounce [m][n]

  const int tid = threadIdx.x, lane = tid & 31, wave = tid >> 5;
  const int lm = lane & 15, hi = lane >> 4, kb = hi * 8;
  const int h = blockIdx.y, b = blockIdx.z;
  const int q0 = blockIdx.x * 128 + wave * 16;
  const size_t hbase = (size_t)(b * T) * 2048 + h * 128;

  // Q fragments (16 rows x 128 d), straight bf16 b128 loads.
  v16bf qa[4];
  {
    const __bf16* qp = Q + hbase + (size_t)(q0 + lm) * 2048;
#pragma unroll
    for (int kc = 0; kc < 4; ++kc) {
      U16x t;
      t.q[0] = *(const uint4*)(qp + kc * 32 + kb);
      t.q[1] = *(const uint4*)(qp + kc * 32 + kb + 16);
      qa[kc] = t.v;
    }
  }

  v8f yacc[8] = {};
  float m[8], l[8];
#pragma unroll
  for (int v = 0; v < 8; ++v) { m[v] = -1e30f; l[v] = 0.0f; }

  const int jmax = blockIdx.x * 128 + 128;
  const int stok = tid >> 3, sdst = (tid & 7) * 16;
  const float SC = 11.313708498984761f;  // sqrt(128)

#if USE_TDM
  const unsigned kt_off = (unsigned)(size_t)(void*)kt;
#endif

  for (int j0 = 0; j0 < jmax; j0 += 32) {
    __syncthreads();
#if USE_TDM
    if (wave == 0) {
      tdm_load_2d(kt_off, K + hbase + (size_t)j0 * 2048, 128u, 32u, 2048u,
                  5u, 3u);
      __builtin_amdgcn_s_wait_tensorcnt((short)0);
    }
#else
    {
      const __bf16* kp = K + hbase + (size_t)(j0 + stok) * 2048 + sdst;
      *(uint4*)&kt[stok * KTS + sdst]     = *(const uint4*)kp;
      *(uint4*)&kt[stok * KTS + sdst + 8] = *(const uint4*)(kp + 8);
    }
#endif
    // V tile, transposed into LDS (manual; TDM cannot transpose).
    {
      const __bf16* vp = V + hbase + (size_t)(j0 + stok) * 2048 + sdst;
      U16x t;
      t.q[0] = *(const uint4*)vp;
      t.q[1] = *(const uint4*)(vp + 8);
#pragma unroll
      for (int i = 0; i < 16; ++i) vt[(sdst + i) * VTS + stok] = t.u[i];
    }
    __syncthreads();

    // S = Q K^T, two 16-column halves
    v8f s0 = {}, s1 = {};
#pragma unroll
    for (int kc = 0; kc < 4; ++kc) {
      int kk = kc * 32 + kb;
      U16x b0, b1;
      b0.q[0] = *(const uint4*)&kt[lm * KTS + kk];
      b0.q[1] = *(const uint4*)&kt[lm * KTS + kk + 16];
      b1.q[0] = *(const uint4*)&kt[(lm + 16) * KTS + kk];
      b1.q[1] = *(const uint4*)&kt[(lm + 16) * KTS + kk + 16];
      s0 = wmma_bf16(qa[kc], b0.v, s0);
      s1 = wmma_bf16(qa[kc], b1.v, s1);
    }

    // online softmax (rows: v + 8*hi; cols: j0 + lm (+16))
    float alpha[8], p0[8], p1[8];
#pragma unroll
    for (int v = 0; v < 8; ++v) {
      int row = q0 + v + 8 * hi;
      float a0 = s0[v] * SC, a1 = s1[v] * SC;
      if (j0 + lm      > row) a0 = -1e30f;
      if (j0 + 16 + lm > row) a1 = -1e30f;
      float rm = redmax16(fmaxf(a0, a1));
      float mn = fmaxf(m[v], rm);
      float al = __expf(m[v] - mn);
      float e0 = __expf(a0 - mn), e1 = __expf(a1 - mn);
      float rs = redsum16(e0 + e1);
      l[v] = l[v] * al + rs;
      m[v] = mn;
      alpha[v] = al; p0[v] = e0; p1[v] = e1;
    }
#pragma unroll
    for (int f = 0; f < 8; ++f)
#pragma unroll
      for (int v = 0; v < 8; ++v) yacc[f][v] *= alpha[v];

    // P: accumulator layout -> A-fragment layout via per-wave LDS bounce
    __bf16* pw = &pb[wave * 16 * VTS];
#pragma unroll
    for (int v = 0; v < 8; ++v) {
      int r = v + 8 * hi;
      pw[r * VTS + lm]      = (__bf16)p0[v];
      pw[r * VTS + lm + 16] = (__bf16)p1[v];
    }
    U16x pa;
    pa.q[0] = *(const uint4*)&pw[lm * VTS + kb];
    pa.q[1] = *(const uint4*)&pw[lm * VTS + kb + 16];

    // Y += P V
#pragma unroll
    for (int nf = 0; nf < 8; ++nf) {
      int drow = nf * 16 + lm;
      U16x bv;
      bv.q[0] = *(const uint4*)&vt[drow * VTS + kb];
      bv.q[1] = *(const uint4*)&vt[drow * VTS + kb + 16];
      yacc[nf] = wmma_bf16(pa.v, bv.v, yacc[nf]);
    }
  }

#pragma unroll
  for (int nf = 0; nf < 8; ++nf)
#pragma unroll
    for (int v = 0; v < 8; ++v) {
      int row = q0 + v + 8 * hi;
      int col = nf * 16 + lm;
      Y[hbase + (size_t)row * 2048 + col] = (__bf16)(yacc[nf][v] / l[v]);
    }
}

// ---------------------------------------------------------------------------
extern "C" void kernel_launch(void* const* d_in, const int* in_sizes, int n_in,
                              void* d_out, int out_size, void* d_ws, size_t ws_size,
                              hipStream_t stream) {
  const float* x  = (const float*)d_in[0];
  const float* Wq = (const float*)d_in[1];
  const float* Wk = (const float*)d_in[2];
  const float* Wv = (const float*)d_in[3];
  const float* Wo = (const float*)d_in[4];
  const float* qs = (const float*)d_in[5];

  const int B = 2, T = 2048, C = 2048;
  const int M = B * T;                       // 4096
  const size_t asz = (size_t)M * C;          // 8,388,608 (activations)
  const size_t wsz = (size_t)C * C;          // 4,194,304 (weights)

  __bf16* bp  = (__bf16*)d_ws;               // ~112 MiB bf16 scratch
  __bf16* xb  = bp;            bp += asz;
  __bf16* wqb = bp;            bp += wsz;
  __bf16* wkb = bp;            bp += wsz;
  __bf16* wvb = bp;            bp += wsz;
  __bf16* wob = bp;            bp += wsz;
  __bf16* qb  = bp;            bp += asz;
  __bf16* kbf = bp;            bp += asz;
  __bf16* vb  = bp;            bp += asz;
  __bf16* yb  = bp;            bp += asz;

  cvt_f32_bf16<<<(int)(asz / 1024), 256, 0, stream>>>(x,  xb,  (int)asz);
  cvt_f32_bf16<<<(int)(wsz / 1024), 256, 0, stream>>>(Wq, wqb, (int)wsz);
  cvt_f32_bf16<<<(int)(wsz / 1024), 256, 0, stream>>>(Wk, wkb, (int)wsz);
  cvt_f32_bf16<<<(int)(wsz / 1024), 256, 0, stream>>>(Wv, wvb, (int)wsz);
  cvt_f32_bf16<<<(int)(wsz / 1024), 256, 0, stream>>>(Wo, wob, (int)wsz);

  dim3 gg(C / 128, M / 128);                 // (16, 32)
  gemm_nt_bf16<false><<<gg, 256, 0, stream>>>(xb, wqb, qb);
  gemm_nt_bf16<false><<<gg, 256, 0, stream>>>(xb, wkb, kbf);
  gemm_nt_bf16<false><<<gg, 256, 0, stream>>>(xb, wvb, vb);

  normrope<<<dim3(16, T, B), 128, 0, stream>>>(qb, kbf, qs, T);

  flash_attn<<<dim3(T / 128, 16, B), 256, 0, stream>>>(qb, kbf, vb, yb, T);

  gemm_nt_bf16<true><<<gg, 256, 0, stream>>>(yb, wob, (float*)d_out);
}
